// eComfEquivariantConvLayer_32366873542776
// MI455X (gfx1250) — matrix operand order repaired and checked by reading, not compiled, not run on hardware
//
#include <hip/hip_runtime.h>
#include <cmath>

typedef __attribute__((ext_vector_type(16))) _Float16 v16h;
typedef __attribute__((ext_vector_type(8)))  _Float16 h8;
typedef __attribute__((ext_vector_type(8)))  float    v8f;

#define SQRT3f      1.7320508075688772f
#define INV_SQRT3f  0.5773502691896258f
#define INV_SQRT2f  0.7071067811865476f
#define INV_DENOM   (1.0f/32.0f)

#define WAVES 8     // waves per block in edge kernel
#define WST   72    // padded LDS stride (halfs) for weight rows
#define HST   72    // padded LDS stride (halfs) for hidden rows

static __device__ __forceinline__ v8f wmma_f16(v16h a, v16h b, v8f c) {
  // D = A(16x32 f16) * B(32x16 f16) + C(16x16 f32)
  return __builtin_amdgcn_wmma_f32_16x16x32_f16(false, a, false, b, (short)0, c,
                                                false, false);
}
static __device__ __forceinline__ v16h cat16(h8 lo, h8 hi) {
  return __builtin_shufflevector(lo, hi, 0,1,2,3,4,5,6,7,8,9,10,11,12,13,14,15);
}
static __device__ __forceinline__ v16h lds_ld16(const _Float16* p) {
  // two ds_load_b128
  return cat16(*(const h8*)p, *(const h8*)(p + 16));
}
static __device__ __forceinline__ h8 cvt8(const float* p) {
  float4 x0 = *(const float4*)p;
  float4 x1 = *(const float4*)(p + 4);
  h8 r = { (_Float16)x0.x,(_Float16)x0.y,(_Float16)x0.z,(_Float16)x0.w,
           (_Float16)x1.x,(_Float16)x1.y,(_Float16)x1.z,(_Float16)x1.w };
  return r;
}
static __device__ __forceinline__ void fadd(float* p, float v) {
  __hip_atomic_fetch_add(p, v, __ATOMIC_RELAXED, __HIP_MEMORY_SCOPE_AGENT);
}

// ---------------------------------------------------------------- zero agg
__global__ void zero_f32_kernel(float* __restrict__ p, int n) {
  int i = blockIdx.x * blockDim.x + threadIdx.x;
  if (i < n) p[i] = 0.0f;
}

// ------------------------------------------------- per-node precompute (VALU)
// s_nl = s @ W_nl0 ; v_nl = einsum(v, W_nl1) ; skip_s = s @ W_skip0 ;
// skip_v = einsum(v, W_skip1)
__global__ __launch_bounds__(160) void node_pre_kernel(
    const float* __restrict__ node_feature,  // N x 160
    const float* __restrict__ W_skip0,       // 64 x 96
    const float* __restrict__ W_skip1,       // 32 x 32
    const float* __restrict__ W_nl0,         // 64 x 64
    const float* __restrict__ W_nl1,         // 32 x 32
    float* __restrict__ s_nl,                // N x 64
    float* __restrict__ v_nl,                // N x 96
    float* __restrict__ skip_s,              // N x 96
    float* __restrict__ skip_v,              // N x 96
    int N) {
  __shared__ float nf[160];
  const int n = blockIdx.x;
  if (n >= N) return;
  const int j = threadIdx.x;
  nf[j] = node_feature[(size_t)n * 160 + j];
  __syncthreads();
  for (int task = j; task < 352; task += 160) {
    if (task < 64) {                       // s_nl
      const int c = task;
      float acc = 0.f;
      #pragma unroll 8
      for (int k = 0; k < 64; ++k) acc += nf[k] * W_nl0[k * 64 + c];
      s_nl[(size_t)n * 64 + c] = acc;
    } else if (task < 160) {               // skip_s
      const int c = task - 64;
      float acc = 0.f;
      #pragma unroll 8
      for (int k = 0; k < 64; ++k) acc += nf[k] * W_skip0[k * 96 + c];
      skip_s[(size_t)n * 96 + c] = acc;
    } else if (task < 256) {               // v_nl
      const int q = task - 160, c = q / 3, i = q % 3;
      float acc = 0.f;
      #pragma unroll 8
      for (int m = 0; m < 32; ++m) acc += nf[64 + 3 * m + i] * W_nl1[m * 32 + c];
      v_nl[(size_t)n * 96 + q] = acc;
    } else {                               // skip_v
      const int q = task - 256, c = q / 3, i = q % 3;
      float acc = 0.f;
      #pragma unroll 8
      for (int m = 0; m < 32; ++m) acc += nf[64 + 3 * m + i] * W_skip1[m * 32 + c];
      skip_v[(size_t)n * 96 + q] = acc;
    }
  }
}

// ------------------------------------------------- edge kernel (WMMA + scatter)
__global__ __launch_bounds__(32 * WAVES) void edge_kernel(
    const float* __restrict__ edge_attr,     // E x 3
    const float* __restrict__ edge_feature,  // E x 64
    const float* __restrict__ W_fc1,         // 64 x 64
    const float* __restrict__ W_fc2,         // 64 x 224
    const int*   __restrict__ edge_index,    // 2 x E
    const float* __restrict__ s_nl,          // N x 64
    const float* __restrict__ v_nl,          // N x 96
    float* __restrict__ agg,                 // N x 480 (atomic accum)
    int E) {
  // Transposed weights in LDS: Wt[n][k] (f16), padded rows to avoid conflicts.
  __shared__ _Float16 fc1_t[64 * WST];
  __shared__ _Float16 fc2_t[224 * WST];
  __shared__ _Float16 hidden[WAVES][16 * HST];  // [m][h] per wave
  __shared__ int   m_src[WAVES][16];
  __shared__ int   m_dst[WAVES][16];
  __shared__ float m_sh[WAVES][16][3];

  const int tid = threadIdx.x;
  for (int i = tid; i < 64 * 64; i += 32 * WAVES) {
    const int k = i >> 6, n = i & 63;
    fc1_t[n * WST + k] = (_Float16)W_fc1[i];
  }
  for (int i = tid; i < 64 * 224; i += 32 * WAVES) {
    const int k = i / 224, n = i % 224;
    fc2_t[n * WST + k] = (_Float16)W_fc2[i];
  }
  __syncthreads();

  const int wave = tid >> 5;
  const int lane = tid & 31;
  const int half = lane >> 4;   // A: K-offset 8*half ; B: K-offset 8*half
  const int l16  = lane & 15;   // A: row ; B/D: column

  const int tile = blockIdx.x * WAVES + wave;
  const int base = tile * 16;
  if (base >= E) return;  // whole-wave exit; no block syncs after this point

  // ---- stage per-tile edge metadata (src, dst, sh1 = sqrt(3)*rhat)
  if (lane < 16) {
    const int e = base + lane;
    if (e < E) {
      m_src[wave][lane] = edge_index[e];
      m_dst[wave][lane] = edge_index[E + e];
      const float ax = edge_attr[3 * e + 0];
      const float ay = edge_attr[3 * e + 1];
      const float az = edge_attr[3 * e + 2];
      const float s = SQRT3f / sqrtf(ax * ax + ay * ay + az * az);
      m_sh[wave][lane][0] = ax * s;
      m_sh[wave][lane][1] = ay * s;
      m_sh[wave][lane][2] = az * s;
    } else {
      m_src[wave][lane] = 0; m_dst[wave][lane] = 0;
      m_sh[wave][lane][0] = 0.f; m_sh[wave][lane][1] = 0.f; m_sh[wave][lane][2] = 0.f;
    }
  }

  // ---- A fragments (edge_feature, f32 -> f16), ISA 16x32 f16 A layout:
  // lane row = l16; elems 0..7 -> K = 8*half + e, elems 8..15 -> K = 16 + 8*half + (e-8)
  v16h a0, a1;
  {
    const int e = base + l16;
    if (e < E) {
      const float* row = edge_feature + (size_t)e * 64;
      a0 = cat16(cvt8(row +  0 + 8 * half), cvt8(row + 16 + 8 * half));
      a1 = cat16(cvt8(row + 32 + 8 * half), cvt8(row + 48 + 8 * half));
    } else {
      h8 z = {};
      a0 = cat16(z, z);
      a1 = cat16(z, z);
    }
  }

  // ---- fc1: hidden = relu(x @ W_fc1), 4 N-tiles x 2 K-steps
  #pragma unroll
  for (int t = 0; t < 4; ++t) {
    v8f acc = {};
    const _Float16* bp = &fc1_t[(16 * t + l16) * WST + 8 * half];
    acc = wmma_f16(a0, lds_ld16(bp), acc);
    acc = wmma_f16(a1, lds_ld16(bp + 32), acc);
    const int ch = 16 * t + l16;
    #pragma unroll
    for (int r = 0; r < 8; ++r) {
      const int m = r + 8 * half;
      float x = acc[r];
      x = x > 0.f ? x : 0.f;
      hidden[wave][m * HST + ch] = (_Float16)x;
    }
  }
  // all DS writes from this wave complete before cross-lane reads below
  asm volatile("s_wait_dscnt 0" ::: "memory");

  // ---- repack hidden into A fragments
  {
    const _Float16* hrow = &hidden[wave][l16 * HST];
    a0 = cat16(*(const h8*)(hrow +      8 * half), *(const h8*)(hrow + 16 + 8 * half));
    a1 = cat16(*(const h8*)(hrow + 32 + 8 * half), *(const h8*)(hrow + 48 + 8 * half));
  }

  // ---- fc2 + fused tensor-product epilogue.
  // D fragment: lane owns channel ch = 16*t + l16 for edges m = r + 8*half.
  #pragma unroll
  for (int t = 0; t < 14; ++t) {
    v8f acc = {};
    const _Float16* bp = &fc2_t[(16 * t + l16) * WST + 8 * half];
    acc = wmma_f16(a0, lds_ld16(bp), acc);
    acc = wmma_f16(a1, lds_ld16(bp + 32), acc);
    const int ch = 16 * t + l16;
    #pragma unroll
    for (int r = 0; r < 8; ++r) {
      const int m = r + 8 * half;
      const int e = base + m;
      if (e >= E) continue;
      const int src = m_src[wave][m];
      const int dst = m_dst[wave][m];
      const float s0 = m_sh[wave][m][0];
      const float s1 = m_sh[wave][m][1];
      const float s2 = m_sh[wave][m][2];
      const float w = acc[r] * INV_DENOM;
      float* ag = agg + (size_t)src * 480;
      if (t < 4) {                 // w0 * xs -> agg[0:64)
        const float xs = s_nl[(size_t)dst * 64 + ch];
        fadd(ag + ch, w * xs);
      } else if (t < 8) {          // (w1*xs) x sh1 -> agg[96 + 3c + i)
        const int c = ch - 64;
        const float v = w * s_nl[(size_t)dst * 64 + c];
        fadd(ag + 96 + 3 * c + 0, v * s0);
        fadd(ag + 96 + 3 * c + 1, v * s1);
        fadd(ag + 96 + 3 * c + 2, v * s2);
      } else if (t < 10) {         // w2 * xv -> agg[288 + 3c + i)
        const int c = ch - 128;
        const float* xv = v_nl + (size_t)dst * 96 + 3 * c;
        fadd(ag + 288 + 3 * c + 0, w * xv[0]);
        fadd(ag + 288 + 3 * c + 1, w * xv[1]);
        fadd(ag + 288 + 3 * c + 2, w * xv[2]);
      } else if (t < 12) {         // w3 * dot(xv, sh1)/sqrt3 -> agg[64 + c)
        const int c = ch - 160;
        const float* xv = v_nl + (size_t)dst * 96 + 3 * c;
        const float dotv = xv[0] * s0 + xv[1] * s1 + xv[2] * s2;
        fadd(ag + 64 + c, w * dotv * INV_SQRT3f);
      } else {                     // w4 * cross(xv, sh1)/sqrt2 -> agg[384 + 3c + i)
        const int c = ch - 192;
        const float* xv = v_nl + (size_t)dst * 96 + 3 * c;
        const float cx = xv[1] * s2 - xv[2] * s1;
        const float cy = xv[2] * s0 - xv[0] * s2;
        const float cz = xv[0] * s1 - xv[1] * s0;
        fadd(ag + 384 + 3 * c + 0, w * cx * INV_SQRT2f);
        fadd(ag + 384 + 3 * c + 1, w * cy * INV_SQRT2f);
        fadd(ag + 384 + 3 * c + 2, w * cz * INV_SQRT2f);
      }
    }
  }
}

// ------------------------------------------------- per-node output (VALU)
__global__ __launch_bounds__(96) void node_out_kernel(
    const float* __restrict__ agg,       // N x 480 (already / 32)
    const float* __restrict__ W_out_s,   // 96 x 96
    const float* __restrict__ W_out_v,   // 128 x 32
    const float* __restrict__ skip_s,    // N x 96
    const float* __restrict__ skip_v,    // N x 96
    float* __restrict__ out,             // N x 160
    int N) {
  __shared__ float ar[480];
  __shared__ float sOut[96];
  const int n = blockIdx.x;
  if (n >= N) return;
  const int j = threadIdx.x;
  for (int i = j; i < 480; i += 96) ar[i] = agg[(size_t)n * 480 + i];
  __syncthreads();

  // out_s[j] = agg_s . W_out_s[:,j] + skip_s
  float accS = skip_s[(size_t)n * 96 + j];
  #pragma unroll 8
  for (int k = 0; k < 96; ++k) accS += ar[k] * W_out_s[k * 96 + j];
  sOut[j] = accS;

  // out_v component j (c = j/3, i = j%3)
  const int c = j / 3, i = j % 3;
  float accV = skip_v[(size_t)n * 96 + j];
  #pragma unroll 8
  for (int m = 0; m < 128; ++m) accV += ar[96 + 3 * m + i] * W_out_v[m * 32 + c];
  __syncthreads();

  const float g = 1.0f / (1.0f + expf(-sOut[64 + c]));   // sigmoid gate
  out[(size_t)n * 160 + 64 + j] = g * accV;
  if (j < 64) {
    const float x = sOut[j];
    out[(size_t)n * 160 + j] = x / (1.0f + expf(-x));    // silu
  }
}

// ---------------------------------------------------------------- launcher
extern "C" void kernel_launch(void* const* d_in, const int* in_sizes, int n_in,
                              void* d_out, int out_size, void* d_ws, size_t ws_size,
                              hipStream_t stream) {
  const float* node_feature = (const float*)d_in[0];
  const float* edge_attr    = (const float*)d_in[1];
  const float* edge_feature = (const float*)d_in[2];
  const float* W_skip0      = (const float*)d_in[3];
  const float* W_skip1      = (const float*)d_in[4];
  const float* W_nl0        = (const float*)d_in[5];
  const float* W_nl1        = (const float*)d_in[6];
  const float* W_fc1        = (const float*)d_in[7];
  const float* W_fc2        = (const float*)d_in[8];
  const float* W_out_s      = (const float*)d_in[9];
  const float* W_out_v      = (const float*)d_in[10];
  const int*   edge_index   = (const int*)d_in[11];

  const int N = in_sizes[0] / 160;
  const int E = in_sizes[1] / 3;

  float* ws     = (float*)d_ws;
  float* agg    = ws;                          // N*480
  float* s_nl   = agg    + (size_t)N * 480;    // N*64
  float* v_nl   = s_nl   + (size_t)N * 64;     // N*96
  float* skip_s = v_nl   + (size_t)N * 96;     // N*96
  float* skip_v = skip_s + (size_t)N * 96;     // N*96

  const int zn = N * 480;
  zero_f32_kernel<<<(zn + 255) / 256, 256, 0, stream>>>(agg, zn);

  node_pre_kernel<<<N, 160, 0, stream>>>(node_feature, W_skip0, W_skip1, W_nl0,
                                         W_nl1, s_nl, v_nl, skip_s, skip_v, N);

  const int tiles  = (E + 15) / 16;
  const int blocks = (tiles + WAVES - 1) / WAVES;
  edge_kernel<<<blocks, 32 * WAVES, 0, stream>>>(edge_attr, edge_feature, W_fc1,
                                                 W_fc2, edge_index, s_nl, v_nl,
                                                 agg, E);

  node_out_kernel<<<N, 96, 0, stream>>>(agg, W_out_s, W_out_v, skip_s, skip_v,
                                        (float*)d_out, N);
}
